// VoxelBackBone8xFocal_36919538876803
// MI455X (gfx1250) — matrix operand order repaired and checked by reading, not compiled, not run on hardware
//
#include <hip/hip_runtime.h>
#include <hip/hip_bf16.h>

// ---------------------------------------------------------------------------
// VoxelBackBone8xFocal for MI455X (gfx1250, wave32, WMMA)
// Sparse submanifold 3D conv == gather + GEMM:
//   out[N,cout] = sum_k gather_k(f)[N,cin] @ W[k][cin,cout]
// K = 27*cin, chunked by 32, V_WMMA_F32_16X16X32_BF16 with f32 accumulation.
// Wide gather units (b64/b128 per neighbor), double-buffered LDS A staging,
// b128 A/B fragment loads, 1 barrier per K chunk.
// ---------------------------------------------------------------------------

typedef __attribute__((ext_vector_type(16))) __bf16 v16bf;
typedef __attribute__((ext_vector_type(8)))  float  v8f;
typedef __attribute__((ext_vector_type(8)))  unsigned int v8u;

#define BB 2
#define DD 22
#define HH 400
#define WW 352
#define NVOX 150000
#define IMAP_TOTAL (BB*DD*HH*WW)   // 6,195,200
#define BN_EPS 0.001f

// ---------------- small helpers ----------------
__device__ __forceinline__ unsigned short f2bf(float f) {
  unsigned int u = __float_as_uint(f);
  unsigned int r = u + 0x7FFFu + ((u >> 16) & 1u);   // round-to-nearest-even
  return (unsigned short)(r >> 16);
}

// ---------------- setup kernels ----------------
__global__ void k_fill_imap(int* imap, int total) {
  int t = blockIdx.x * blockDim.x + threadIdx.x;
  if (t < total) imap[t] = -1;
}

__global__ void k_scatter_imap(const int* __restrict__ coords, int* __restrict__ imap, int n) {
  int i = blockIdx.x * blockDim.x + threadIdx.x;
  if (i < n) {
    int b = coords[4*i+0], z = coords[4*i+1], y = coords[4*i+2], x = coords[4*i+3];
    imap[((b*DD + z)*HH + y)*WW + x] = i;
  }
}

__global__ void k_build_nb(const int* __restrict__ coords, const int* __restrict__ imap,
                           int* __restrict__ nb, int n) {
  int t = blockIdx.x * blockDim.x + threadIdx.x;
  if (t >= n * 27) return;
  int i = t / 27, k = t % 27;
  int dz = k / 9 - 1, dy = (k / 3) % 3 - 1, dx = k % 3 - 1;
  int b = coords[4*i+0];
  int z = coords[4*i+1] + dz, y = coords[4*i+2] + dy, x = coords[4*i+3] + dx;
  int r = -1;
  if (z >= 0 && z < DD && y >= 0 && y < HH && x >= 0 && x < WW)
    r = imap[((b*DD + z)*HH + y)*WW + x];
  nb[t] = r;
}

__global__ void k_init_kmaps(int* k27, int* k2) {
  int t = threadIdx.x;
  if (t < 27) k27[t] = t;
  if (t == 0) { k2[0] = 13; k2[1] = 22; }   // (0,0,0) -> self col13, (1,0,0) -> col22
}

__global__ void k_cvt_bf16(const float* __restrict__ src, unsigned short* __restrict__ dst, int total) {
  int t = blockIdx.x * blockDim.x + threadIdx.x;
  if (t < total) dst[t] = f2bf(src[t]);
}

// Pack weights (K27,cin,cout_src) f32 -> bf16 [cout_dst][kpad], K-major per column,
// zero padded in K and in columns (mask conv pads cout 1 -> 16).
__global__ void k_pack_w(const float* __restrict__ w, unsigned short* __restrict__ dst,
                         int K27, int cin, int cout_src, int cout_dst, int kpad) {
  int t = blockIdx.x * blockDim.x + threadIdx.x;
  if (t >= cout_dst * kpad) return;
  int co = t / kpad, kk = t % kpad;
  int k = kk / cin, c = kk % cin;
  float v = 0.0f;
  if (k < K27 && co < cout_src) v = w[((size_t)(k * cin + c)) * cout_src + co];
  dst[(size_t)co * kpad + kk] = f2bf(v);
}

__global__ void k_zero_stats(float* s) { if (threadIdx.x < 256) s[threadIdx.x] = 0.0f; }

// per-channel sum / sumsq over N rows
__global__ void k_stats(const float* __restrict__ x, int n, int C, float* __restrict__ stats) {
  __shared__ float ssum[128], ssq[128];
  int tid = threadIdx.x;
  if (tid < C) { ssum[tid] = 0.0f; ssq[tid] = 0.0f; }
  __syncthreads();
  int c = tid % C;
  int rpb = 256 / C;                       // C in {16,32,64,128}
  float s = 0.0f, q = 0.0f;
  for (int r = blockIdx.x * rpb + tid / C; r < n; r += gridDim.x * rpb) {
    float v = x[(size_t)r * C + c];
    s += v; q += v * v;
  }
  atomicAdd(&ssum[c], s); atomicAdd(&ssq[c], q);
  __syncthreads();
  if (tid < C) { atomicAdd(&stats[c], ssum[tid]); atomicAdd(&stats[C + c], ssq[tid]); }
}

// BN(+ReLU)(+sigmoid-mask multiply) epilogue. Writes bf16 (next layer) or f32 (final).
__global__ void k_bn_apply(const float* __restrict__ x, const float* __restrict__ stats,
                           const float* __restrict__ gamma, const float* __restrict__ beta,
                           const float* __restrict__ mask,      // [N][16] col0, or null
                           unsigned short* __restrict__ obf, float* __restrict__ of32,
                           int n, int C) {
  int t = blockIdx.x * blockDim.x + threadIdx.x;
  if (t >= n * C) return;
  int row = t / C, c = t % C;
  float inv_n = 1.0f / (float)n;
  float mean = stats[c] * inv_n;
  float var  = fmaxf(stats[C + c] * inv_n - mean * mean, 0.0f);
  float y = (x[t] - mean) * rsqrtf(var + BN_EPS) * gamma[c] + beta[c];
  y = fmaxf(y, 0.0f);
  if (mask) {
    float m = mask[(size_t)row * 16];
    y *= 1.0f / (1.0f + expf(-m));
  }
  if (obf) obf[t] = f2bf(y);
  else     of32[t] = y;
}

// ---------------- fused gather + WMMA GEMM ----------------
// 256 threads = 8 waves. Wave w handles output tile (msub = w/NTN, nsub = w%NTN).
// Gather granule: one neighbor row segment of UNIT_DW dwords (b64 for cin=4,
// b128 otherwise) -> 1 nb lookup + 1 wide load + 1 wide LDS store per unit.
// Double-buffered LDS A; per chunk: 2x ds_load_b128 A frag, 2x global b128
// B frag, 1 WMMA, 1 barrier.
template <int CIN, int COUT>
__global__ __launch_bounds__(256) void conv_wmma(
    const unsigned short* __restrict__ fin,     // [N][CIN] bf16
    const unsigned short* __restrict__ wpack,   // [COUT][kpad] bf16, K-major
    const int* __restrict__ nbtab,              // [N][27]
    const int* __restrict__ kmap_g, int nk, int kpad,
    float* __restrict__ out, int n)             // [N][COUT] f32 (pre-BN)
{
  constexpr int NTN  = COUT / 16;      // col tiles per block
  constexpr int NTM  = 8 / NTN;        // row tiles per block
  constexpr int MBLK = 16 * NTM;       // voxels per block
  constexpr int UNIT_DW = (CIN == 4) ? 2 : 4;    // dwords per gather unit
  constexpr int UPR   = 16 / UNIT_DW;            // units per row (16 dwords/row)
  constexpr int UNITS = MBLK * UPR;              // units per chunk
  constexpr int UPT   = (UNITS + 255) / 256;     // units per thread

  __shared__ unsigned int As[2][MBLK * 16];      // [buf][MBLK][16] dwords
  __shared__ int skmap[32];

  int tid  = threadIdx.x;
  if (tid < 32) skmap[tid] = (tid < nk) ? kmap_g[tid] : 0;
  int wave = tid >> 5, lane = tid & 31;
  int lo   = lane & 15;
  bool hi  = (lane >= 16);
  int msub = wave / NTN, nsub = wave % NTN;
  int m0   = blockIdx.x * MBLK;
  int col  = nsub * 16 + lo;
  int chunks = kpad >> 5;

  __syncthreads();   // skmap visible

  uint4 regs[UPT];

  // gather one 32-wide K chunk (MBLK x 32 bf16) into registers, unit = one
  // neighbor-row segment (always inside a single stencil tap)
  auto gather = [&](int ch) {
#pragma unroll
    for (int p = 0; p < UPT; ++p) {
      int u = tid + p * 256;
      uint4 val = {0u, 0u, 0u, 0u};
      if (UNITS >= 256 || u < UNITS) {
        int m = u / UPR, s = u % UPR;            // row, segment-in-row
        int kg = (ch << 5) + s * (UNIT_DW * 2);  // bf16-K of segment start
        int koff = kg / CIN, c = kg % CIN;
        int v = m0 + m;
        if (v < n && koff < nk) {
          int nbr = nbtab[v * 27 + skmap[koff]];
          if (nbr >= 0) {
            const unsigned short* src = fin + (size_t)nbr * CIN + c;
            if (UNIT_DW == 2) {
              uint2 t2 = *(const uint2*)src;
              val.x = t2.x; val.y = t2.y;
            } else {
              val = *(const uint4*)src;
            }
          }
        }
      }
      regs[p] = val;
    }
  };

  // store staged registers into LDS buffer `buf`
  auto stage = [&](int buf) {
#pragma unroll
    for (int p = 0; p < UPT; ++p) {
      int u = tid + p * 256;
      if (UNITS >= 256 || u < UNITS) {
        int m = u / UPR, s = u % UPR;
        unsigned int* dst = &As[buf][m * 16 + s * UNIT_DW];
        if (UNIT_DW == 2) {
          uint2 t2; t2.x = regs[p].x; t2.y = regs[p].y;
          *(uint2*)dst = t2;
        } else {
          *(uint4*)dst = regs[p];
        }
      }
    }
  };

  gather(0);
  stage(0);
  __syncthreads();

  v8f acc = {};
  int arow = (msub * 16 + lo) * 16;
  int kb = hi ? 8 : 0;                         // dword offset of K-base (16 bf16)

#pragma unroll 2
  for (int ch = 0; ch < chunks; ++ch) {
    int cur = ch & 1, nxt = cur ^ 1;

    // issue next chunk's gathers early (overlap with compute below)
    if (ch + 1 < chunks) gather(ch + 1);

    // A contiguous K order (matches B): lane<16 -> K 0..15; lane>=16 -> K 16..31
    // dwords {kb..kb+3} and {kb+4..kb+7}: two ds_load_b128
    uint4 a0 = *(const uint4*)&As[cur][arow + kb];
    uint4 a1 = *(const uint4*)&As[cur][arow + kb + 4];
    // B contiguous K order (matches A): lane<16 col=lane K 0..15; lane>=16 K 16..31
    // 8 consecutive dwords, 16B aligned: two global_load_b128
    const uint4* bp =
        (const uint4*)(wpack + (size_t)col * kpad + (ch << 5) + (hi ? 16 : 0));
    uint4 b0 = bp[0], b1 = bp[1];

    v8u au, bu;
    au[0] = a0.x; au[1] = a0.y; au[2] = a0.z; au[3] = a0.w;
    au[4] = a1.x; au[5] = a1.y; au[6] = a1.z; au[7] = a1.w;
    bu[0] = b0.x; bu[1] = b0.y; bu[2] = b0.z; bu[3] = b0.w;
    bu[4] = b1.x; bu[5] = b1.y; bu[6] = b1.z; bu[7] = b1.w;

    acc = __builtin_amdgcn_wmma_f32_16x16x32_bf16(
        false, __builtin_bit_cast(v16bf, au),
        false, __builtin_bit_cast(v16bf, bu),
        (short)0, acc, false, false);

    // store next chunk into the other buffer, then one barrier
    if (ch + 1 < chunks) stage(nxt);
    __syncthreads();
  }

  // D layout: vgpr r: lane<16 -> D[r][lane]; lane>=16 -> D[r+8][lane-16]
  int rbase = m0 + msub * 16 + (hi ? 8 : 0);
#pragma unroll
  for (int r = 0; r < 8; ++r) {
    int row = rbase + r;
    if (row < n) out[(size_t)row * COUT + col] = acc[r];
  }
}

template <int CIN, int COUT>
static void launch_conv(const unsigned short* fin, const unsigned short* wp,
                        const int* nb, const int* km, int nk, int kpad,
                        float* out, int n, hipStream_t s) {
  constexpr int MBLK = 16 * (8 / (COUT / 16));
  dim3 grid((n + MBLK - 1) / MBLK);
  conv_wmma<CIN, COUT><<<grid, 256, 0, s>>>(fin, wp, nb, km, nk, kpad, out, n);
}

static void conv_dispatch(int CIN, int COUT,
                          const unsigned short* fin, const unsigned short* wp,
                          const int* nb, const int* km, int nk, int kpad,
                          float* out, int n, hipStream_t s) {
  if      (CIN ==  4 && COUT ==  16) launch_conv< 4, 16>(fin, wp, nb, km, nk, kpad, out, n, s);
  else if (CIN == 16 && COUT ==  16) launch_conv<16, 16>(fin, wp, nb, km, nk, kpad, out, n, s);
  else if (CIN == 16 && COUT ==  32) launch_conv<16, 32>(fin, wp, nb, km, nk, kpad, out, n, s);
  else if (CIN == 32 && COUT ==  32) launch_conv<32, 32>(fin, wp, nb, km, nk, kpad, out, n, s);
  else if (CIN == 32 && COUT ==  16) launch_conv<32, 16>(fin, wp, nb, km, nk, kpad, out, n, s);
  else if (CIN == 32 && COUT ==  64) launch_conv<32, 64>(fin, wp, nb, km, nk, kpad, out, n, s);
  else if (CIN == 64 && COUT ==  64) launch_conv<64, 64>(fin, wp, nb, km, nk, kpad, out, n, s);
  else if (CIN == 64 && COUT ==  16) launch_conv<64, 16>(fin, wp, nb, km, nk, kpad, out, n, s);
  else if (CIN == 64 && COUT == 128) launch_conv<64,128>(fin, wp, nb, km, nk, kpad, out, n, s);
}

// ---------------------------------------------------------------------------
extern "C" void kernel_launch(void* const* d_in, const int* in_sizes, int n_in,
                              void* d_out, int out_size, void* d_ws, size_t ws_size,
                              hipStream_t stream) {
  (void)n_in; (void)out_size; (void)ws_size;
  const float* vf     = (const float*)d_in[0];
  const int*   coords = (const int*)d_in[1];

  // layers: 0=in 1=c1 2=f1 3=c2a 4=c2b 5=c2c 6=f2 7=c3a 8=c3b 9=c3c 10=f3 11=c4a 12=c4b 13=c4c 14=out
  const int NL = 15;
  int  cins[NL]  = { 4,16,16,16,32,32,32,32,64,64,64,64,64,64,64};
  int  couts[NL] = {16,16,16,32,32,32,32,64,64,64,64,64,64,64,128};
  int  k27s[NL]  = {27,27,27,27,27,27,27,27,27,27,27,27,27,27, 2};
  bool foc[NL]   = { 0, 0, 1, 0, 0, 0, 1, 0, 0, 0, 1, 0, 0, 0, 0};
  const float *Wv[NL], *Gv[NL], *Bv[NL], *WMv[NL];
  for (int i = 0; i < NL; ++i) WMv[i] = nullptr;

  auto F = [&](int i) { return (const float*)d_in[i]; };
  if (in_sizes[2] == 16) {
    // pytree-sorted dict flattening: layers sorted, leaves b,g,w[,wm]
    int order[NL] = {1,3,4,5,7,8,9,11,12,13,2,6,10,0,14};
    int idx = 2;
    for (int t = 0; t < NL; ++t) {
      int L = order[t];
      Bv[L] = F(idx++); Gv[L] = F(idx++); Wv[L] = F(idx++);
      if (foc[L]) WMv[L] = F(idx++);
    }
  } else {
    // insertion-order flattening: layers in creation order, leaves w,g,b[,wm]
    int idx = 2;
    for (int L = 0; L < NL; ++L) {
      Wv[L] = F(idx++); Gv[L] = F(idx++); Bv[L] = F(idx++);
      if (foc[L]) WMv[L] = F(idx++);
    }
  }

  // workspace carve-up (~130 MB)
  char* base = (char*)d_ws;
  size_t off = 0;
  auto take = [&](size_t bytes) -> void* {
    void* p = base + off;
    off += (bytes + 255) & ~(size_t)255;
    return p;
  };
  int*            imap    = (int*)take((size_t)IMAP_TOTAL * 4);
  int*            nb27    = (int*)take((size_t)NVOX * 27 * 4);
  unsigned short* featA   = (unsigned short*)take((size_t)NVOX * 64 * 2);
  unsigned short* featB   = (unsigned short*)take((size_t)NVOX * 64 * 2);
  float*          convout = (float*)take((size_t)NVOX * 64 * 4);
  float*          maskout = (float*)take((size_t)NVOX * 16 * 4);
  float*          stats   = (float*)take(1024);
  int*            kmap27  = (int*)take(128);
  int*            kmap2   = (int*)take(128);
  unsigned short* wbuf    = (unsigned short*)take(2ull * 1024 * 1024);

  // ---- one-time-per-call setup (graph-capture safe, deterministic) ----
  k_fill_imap   <<<(IMAP_TOTAL + 255) / 256, 256, 0, stream>>>(imap, IMAP_TOTAL);
  k_scatter_imap<<<(NVOX + 255) / 256,       256, 0, stream>>>(coords, imap, NVOX);
  k_build_nb    <<<(NVOX * 27 + 255) / 256,  256, 0, stream>>>(coords, imap, nb27, NVOX);
  k_init_kmaps  <<<1, 32, 0, stream>>>(kmap27, kmap2);
  k_cvt_bf16    <<<(NVOX * 4 + 255) / 256,   256, 0, stream>>>(vf, featA, NVOX * 4);

  // pack weights (bf16, [cout][kpad] K-major)
  unsigned short* wpk[NL];
  unsigned short* wpkm[NL];
  int kpads[NL];
  {
    unsigned short* cur = wbuf;
    for (int L = 0; L < NL; ++L) {
      int kpad = ((k27s[L] * cins[L] + 31) / 32) * 32;
      kpads[L] = kpad;
      wpk[L] = cur; cur += (size_t)couts[L] * kpad;
      int tot = couts[L] * kpad;
      k_pack_w<<<(tot + 255) / 256, 256, 0, stream>>>(
          Wv[L], wpk[L], k27s[L], cins[L], couts[L], couts[L], kpad);
      wpkm[L] = nullptr;
      if (foc[L]) {
        wpkm[L] = cur; cur += (size_t)16 * kpad;
        int tm = 16 * kpad;
        k_pack_w<<<(tm + 255) / 256, 256, 0, stream>>>(
            WMv[L], wpkm[L], 27, cins[L], 1, 16, kpad);
      }
    }
  }

  // ---- layer pipeline ----
  unsigned short* curf = featA;
  unsigned short* nxtf = featB;
  for (int L = 0; L < NL; ++L) {
    int CIN = cins[L], COUT = couts[L], nk = k27s[L], kpad = kpads[L];
    const int* km = (nk == 27) ? kmap27 : kmap2;
    float* cOut = (L == NL - 1) ? (float*)d_out : convout;

    conv_dispatch(CIN, COUT, curf, wpk[L], nb27, km, nk, kpad, cOut, NVOX, stream);
    if (foc[L])
      conv_dispatch(CIN, 16, curf, wpkm[L], nb27, kmap27, 27, kpad, maskout, NVOX, stream);

    k_zero_stats<<<1, 256, 0, stream>>>(stats);
    k_stats<<<512, 256, 0, stream>>>(cOut, NVOX, COUT, stats);

    int tot = NVOX * COUT;
    if (L == NL - 1) {
      k_bn_apply<<<(tot + 255) / 256, 256, 0, stream>>>(
          cOut, stats, Gv[L], Bv[L], nullptr, nullptr, (float*)d_out, NVOX, COUT);
    } else {
      k_bn_apply<<<(tot + 255) / 256, 256, 0, stream>>>(
          cOut, stats, Gv[L], Bv[L], foc[L] ? maskout : nullptr, nxtf, nullptr, NVOX, COUT);
      unsigned short* t = curf; curf = nxtf; nxtf = t;
    }
  }
}
